// Toy_MultiHeadAttention_58480274703169
// MI455X (gfx1250) — compile-verified, matrix operand order checked by reading, not run on hardware
//
#include <hip/hip_runtime.h>
#include <hip/hip_bf16.h>

// Problem constants (reference: B=2, L=2048, D=1024, H=16, HD=64)
#define BB   2
#define LL   2048
#define DD   1024
#define HH   16
#define HDD  64

typedef __attribute__((ext_vector_type(16))) _Float16 v16h;
typedef __attribute__((ext_vector_type(4)))  _Float16 v4h;
typedef __attribute__((ext_vector_type(8)))  float    v8f;
typedef __attribute__((ext_vector_type(4)))  float    v4f;

// ---------------------------------------------------------------------------
// WMMA 16x16x32 f16 fragment helpers (wave32).
//
// A-matrix 16x32 (MxK), 16-bit (ISA 7.12.2):
//   lane l<16 : row M=l,    halves 0..7 = K 0..7,  halves 8..15 = K 16..23
//   lane l>=16: row M=l-16, halves 0..7 = K 8..15, halves 8..15 = K 24..31
// B-matrix 32x16 (KxN) lane-symmetric: lane owns column N=lane&15 with the
// same K striping. C/D 16x16 f32: VGPR j holds M=j (lanes 0-15) / M=j+8
// (lanes 16-31), N=lane&15.
// ---------------------------------------------------------------------------

__device__ inline v16h frag_kcontig(const _Float16* __restrict__ base, int ld,
                                    int r0, int k0, int lane) {
  const int r  = r0 + (lane & 15);
  const int kb = k0 + ((lane & 16) ? 8 : 0);
  const _Float16* p = base + (size_t)r * ld + kb;
  v16h f;
#pragma unroll
  for (int j = 0; j < 8; ++j) { f[j] = p[j]; f[j + 8] = p[16 + j]; }
  return f;
}

// B fragment where B[k][n] is stored row-major as base[k*ld + n]
// (contraction dimension strided by ld), e.g. V in P @ V.
__device__ inline v16h frag_kstrided(const _Float16* __restrict__ base, int ld,
                                     int k0, int n0, int lane) {
  const int n  = n0 + (lane & 15);
  const int kb = k0 + ((lane & 16) ? 8 : 0);
  const _Float16* p = base + (size_t)kb * ld + n;
  v16h f;
#pragma unroll
  for (int j = 0; j < 8; ++j) {
    f[j]     = p[(size_t)j * ld];
    f[j + 8] = p[(size_t)(16 + j) * ld];
  }
  return f;
}

__device__ inline v8f wmma16(v16h a, v16h b, v8f c) {
  return __builtin_amdgcn_wmma_f32_16x16x32_f16(
      /*neg_a=*/false, a, /*neg_b=*/false, b,
      /*c_mod=*/(short)0, c, /*reuse_a=*/false, /*reuse_b=*/false);
}

// ---------------------------------------------------------------------------
// Kernel 1: f32 -> f16 conversion, 4 elements/thread
// ---------------------------------------------------------------------------
__global__ __launch_bounds__(256)
void k_cvt_f16(const float* __restrict__ src, _Float16* __restrict__ dst, int n4) {
  int i = blockIdx.x * blockDim.x + threadIdx.x;
  if (i < n4) {
    v4f s = ((const v4f*)src)[i];
    v4h d;
    d[0] = (_Float16)s[0]; d[1] = (_Float16)s[1];
    d[2] = (_Float16)s[2]; d[3] = (_Float16)s[3];
    ((v4h*)dst)[i] = d;
  }
}

// ---------------------------------------------------------------------------
// Kernel 2: QKV projection, register-blocked 32(M) x 64(N) per wave.
// N-tile (64 wide) aligns exactly with one (which,head) slice since HD=64.
// 8 WMMAs per k-step from 6 fragment loads.  Q pre-scaled by 1/sqrt(HD).
// ---------------------------------------------------------------------------
__global__ __launch_bounds__(32)
void k_qkv_gemm(const _Float16* __restrict__ X, const _Float16* __restrict__ W,
                _Float16* __restrict__ Q, _Float16* __restrict__ K,
                _Float16* __restrict__ V) {
  const int lane = threadIdx.x;
  const int n0 = blockIdx.x * 64;          // 0..3008, multiple of HD
  const int m0 = blockIdx.y * 32;          // 0..4064
  v8f acc[2][4] = {};
  for (int k0 = 0; k0 < DD; k0 += 32) {
    // WGP-scope prefetch one k-step ahead (speculative; harmless at the tail)
    __builtin_prefetch(X + (size_t)(m0 + (lane & 31)) * DD + k0 + 32, 0, 3);
    __builtin_prefetch(W + (size_t)(n0 + (lane & 31)) * DD + k0 + 32, 0, 3);
    const v16h a0 = frag_kcontig(X, DD, m0,      k0, lane);
    const v16h a1 = frag_kcontig(X, DD, m0 + 16, k0, lane);
#pragma unroll
    for (int t = 0; t < 4; ++t) {
      const v16h b = frag_kcontig(W, DD, n0 + t * 16, k0, lane);
      acc[0][t] = wmma16(a0, b, acc[0][t]);
      acc[1][t] = wmma16(a1, b, acc[1][t]);
    }
  }
  // Epilogue: whole N-tile belongs to one (which, h) slice.
  const int which = n0 >> 10;              // 0=q,1=k,2=v
  const int h     = (n0 & (DD - 1)) >> 6;
  _Float16* dst = (which == 0) ? Q : ((which == 1) ? K : V);
  const float scale = (which == 0) ? 0.125f : 1.0f;
  const int hi = (lane & 16) ? 8 : 0;
  const int nn = lane & 15;
#pragma unroll
  for (int mt = 0; mt < 2; ++mt) {
#pragma unroll
    for (int j = 0; j < 8; ++j) {
      const int m  = m0 + mt * 16 + j + hi;
      const int b_ = m >> 11;              // /L
      const int l  = m & (LL - 1);
      _Float16* o = dst + ((((size_t)b_ * HH + h) * LL) + l) * HDD;
#pragma unroll
      for (int t = 0; t < 4; ++t)
        o[t * 16 + nn] = (_Float16)(acc[mt][t][j] * scale);
    }
  }
}

// ---------------------------------------------------------------------------
// Kernel 3: fused causal attention, flash-style, one wave per
// (b, h, 16-query tile).  Keys streamed in chunks of 32.
// Output written to [B*L, D] f16 (implicit head-merge transpose).
// ---------------------------------------------------------------------------
__global__ __launch_bounds__(32)
void k_attn(const _Float16* __restrict__ Q, const _Float16* __restrict__ Km,
            const _Float16* __restrict__ Vm, _Float16* __restrict__ O) {
  __shared__ float    s_sc[16][33];   // score tile, padded
  __shared__ _Float16 s_p[16][32];    // probabilities (f16) for P@V
  __shared__ float    s_corr[16];
  __shared__ float    s_sum[16];

  const int lane = threadIdx.x;
  const int q0 = blockIdx.x * 16;     // query tile base
  const int h  = blockIdx.y;
  const int b  = blockIdx.z;
  const size_t head = ((size_t)b * HH + h) * LL * HDD;
  const _Float16* qb = Q  + head;
  const _Float16* kb = Km + head;
  const _Float16* vb = Vm + head;

  // Q fragments for the whole tile (HD=64 -> two 32-wide k-steps), reused.
  const v16h qf0 = frag_kcontig(qb, HDD, q0, 0,  lane);
  const v16h qf1 = frag_kcontig(qb, HDD, q0, 32, lane);

  v8f acc0 = {}, acc1 = {}, acc2 = {}, acc3 = {};
  float row_max = -__builtin_inff();  // owned by lanes 0..15 (row = lane)
  float row_sum = 0.0f;

  const int nchunks = q0 / 32 + 1;    // causal: keys [0, q0+15]
  const int hi = (lane & 16) ? 8 : 0;
  const int nn = lane & 15;

  for (int ch = 0; ch < nchunks; ++ch) {
    const int kk0 = ch * 32;

    // S = Q . K^T for two 16-key tiles (each: two WMMAs over HD)
    v8f s0 = {}, s1 = {};
    s0 = wmma16(qf0, frag_kcontig(kb, HDD, kk0,      0,  lane), s0);
    s0 = wmma16(qf1, frag_kcontig(kb, HDD, kk0,      32, lane), s0);
    s1 = wmma16(qf0, frag_kcontig(kb, HDD, kk0 + 16, 0,  lane), s1);
    s1 = wmma16(qf1, frag_kcontig(kb, HDD, kk0 + 16, 32, lane), s1);

    // Spill D-layout score tiles to LDS
#pragma unroll
    for (int j = 0; j < 8; ++j) {
      s_sc[j + hi][nn]      = s0[j];
      s_sc[j + hi][nn + 16] = s1[j];
    }
    __syncthreads();

    // Online softmax: lane r < 16 owns query row q0+r
    if (lane < 16) {
      const int r  = lane;
      const int qi = q0 + r;
      float mx = row_max;
#pragma unroll
      for (int cc = 0; cc < 32; ++cc) {
        float s = (kk0 + cc <= qi) ? s_sc[r][cc] : -__builtin_inff();
        s_sc[r][cc] = s;
        mx = fmaxf(mx, s);
      }
      const float corr = __expf(row_max - mx);
      float sum = 0.0f;
#pragma unroll
      for (int cc = 0; cc < 32; ++cc) {
        const float p = __expf(s_sc[r][cc] - mx);
        sum += p;
        s_p[r][cc] = (_Float16)p;
      }
      row_max = mx;
      row_sum = row_sum * corr + sum;
      s_corr[r] = corr;
    }
    __syncthreads();

    // Rescale running output accumulators by per-row correction
#pragma unroll
    for (int j = 0; j < 8; ++j) {
      const float corr = s_corr[j + hi];
      acc0[j] *= corr; acc1[j] *= corr; acc2[j] *= corr; acc3[j] *= corr;
    }

    // O += P . V   (P from LDS as A-fragment; V columns strided by HD)
    const v16h pf = frag_kcontig(&s_p[0][0], 32, 0, 0, lane);
    acc0 = wmma16(pf, frag_kstrided(vb, HDD, kk0, 0,  lane), acc0);
    acc1 = wmma16(pf, frag_kstrided(vb, HDD, kk0, 16, lane), acc1);
    acc2 = wmma16(pf, frag_kstrided(vb, HDD, kk0, 32, lane), acc2);
    acc3 = wmma16(pf, frag_kstrided(vb, HDD, kk0, 48, lane), acc3);
    __syncthreads();   // s_sc / s_p reused next chunk
  }

  if (lane < 16) s_sum[lane] = row_sum;
  __syncthreads();

  // Normalize and store to [B*L, D] f16 at column h*HD + d
#pragma unroll
  for (int j = 0; j < 8; ++j) {
    const int l = q0 + j + hi;
    const float inv = 1.0f / s_sum[j + hi];
    _Float16* o = O + ((size_t)b * LL + l) * DD + h * HDD;
    o[nn]      = (_Float16)(acc0[j] * inv);
    o[nn + 16] = (_Float16)(acc1[j] * inv);
    o[nn + 32] = (_Float16)(acc2[j] * inv);
    o[nn + 48] = (_Float16)(acc3[j] * inv);
  }
}

// ---------------------------------------------------------------------------
// Kernel 4: output projection, register-blocked 32(M) x 64(N) per wave.
// A[4096,1024]f16 x Wout[1024,1024]f16^T + bias -> f32.
// ---------------------------------------------------------------------------
__global__ __launch_bounds__(32)
void k_out_gemm(const _Float16* __restrict__ A, const _Float16* __restrict__ W,
                const float* __restrict__ bias, float* __restrict__ Out) {
  const int lane = threadIdx.x;
  const int n0 = blockIdx.x * 64;
  const int m0 = blockIdx.y * 32;
  v8f acc[2][4] = {};
  for (int k0 = 0; k0 < DD; k0 += 32) {
    __builtin_prefetch(A + (size_t)(m0 + (lane & 31)) * DD + k0 + 32, 0, 3);
    __builtin_prefetch(W + (size_t)(n0 + (lane & 31)) * DD + k0 + 32, 0, 3);
    const v16h a0 = frag_kcontig(A, DD, m0,      k0, lane);
    const v16h a1 = frag_kcontig(A, DD, m0 + 16, k0, lane);
#pragma unroll
    for (int t = 0; t < 4; ++t) {
      const v16h b = frag_kcontig(W, DD, n0 + t * 16, k0, lane);
      acc[0][t] = wmma16(a0, b, acc[0][t]);
      acc[1][t] = wmma16(a1, b, acc[1][t]);
    }
  }
  const int hi = (lane & 16) ? 8 : 0;
  const int nn = lane & 15;
  float bv[4];
#pragma unroll
  for (int t = 0; t < 4; ++t) bv[t] = bias[n0 + t * 16 + nn];
#pragma unroll
  for (int mt = 0; mt < 2; ++mt) {
#pragma unroll
    for (int j = 0; j < 8; ++j) {
      float* o = Out + (size_t)(m0 + mt * 16 + j + hi) * DD + n0;
#pragma unroll
      for (int t = 0; t < 4; ++t)
        o[t * 16 + nn] = acc[mt][t][j] + bv[t];
    }
  }
}

// ---------------------------------------------------------------------------
extern "C" void kernel_launch(void* const* d_in, const int* in_sizes, int n_in,
                              void* d_out, int out_size, void* d_ws, size_t ws_size,
                              hipStream_t stream) {
  (void)in_sizes; (void)n_in; (void)out_size; (void)ws_size;
  const float* x     = (const float*)d_in[0];   // [B,L,D]
  const float* qkv_w = (const float*)d_in[1];   // [3D,D]
  const float* out_w = (const float*)d_in[2];   // [D,D]
  const float* out_b = (const float*)d_in[3];   // [D]
  float* out = (float*)d_out;                    // [B,L,D]

  char* ws = (char*)d_ws;
  const size_t MiB = 1ull << 20;
  _Float16* xh    = (_Float16*)(ws +  0 * MiB); // 4096x1024 f16 (8 MiB)
  _Float16* wqkvh = (_Float16*)(ws +  8 * MiB); // 3072x1024 f16 (6 MiB)
  _Float16* owh   = (_Float16*)(ws + 14 * MiB); // 1024x1024 f16 (2 MiB)
  _Float16* qh    = (_Float16*)(ws + 16 * MiB); // [B,H,L,HD] f16 (8 MiB)
  _Float16* kh    = (_Float16*)(ws + 24 * MiB); // [B,H,L,HD] f16 (8 MiB)
  _Float16* vh    = (_Float16*)(ws + 32 * MiB); // [B,H,L,HD] f16 (8 MiB)
  _Float16* ah    = (_Float16*)(ws + 40 * MiB); // [B*L, D]   f16 (8 MiB)

  const int n_x = BB * LL * DD;        // 4194304
  const int n_w = 3 * DD * DD;         // 3145728
  const int n_o = DD * DD;             // 1048576
  k_cvt_f16<<<dim3(n_x / 4 / 256), dim3(256), 0, stream>>>(x, xh, n_x / 4);
  k_cvt_f16<<<dim3(n_w / 4 / 256), dim3(256), 0, stream>>>(qkv_w, wqkvh, n_w / 4);
  k_cvt_f16<<<dim3(n_o / 4 / 256), dim3(256), 0, stream>>>(out_w, owh, n_o / 4);

  // QKV GEMM: N=3072 in 64-wide tiles (x), M=4096 in 32-wide tiles (y)
  k_qkv_gemm<<<dim3(3 * DD / 64, BB * LL / 32), dim3(32), 0, stream>>>(
      xh, wqkvh, qh, kh, vh);

  // Attention: (q-tiles, heads, batch)
  k_attn<<<dim3(LL / 16, HH, BB), dim3(32), 0, stream>>>(qh, kh, vh, ah);

  // Output GEMM + bias
  k_out_gemm<<<dim3(DD / 64, BB * LL / 32), dim3(32), 0, stream>>>(
      ah, owh, out_b, out);
}
